// AnatomicalSpaceAttention_5952824672905
// MI455X (gfx1250) — matrix operand order repaired and checked by reading, not compiled, not run on hardware
//
#include <hip/hip_runtime.h>
#include <hip/hip_bf16.h>
#include <math.h>

// ---------------------------------------------------------------------------
// AnatomicalSpaceAttention on gfx1250 (MI455X), bf16 WMMA pipeline.
// B=2, C=256, D=H=W=32 (N=32768), S=256, TD=512, NH=8, HD=32.
// ---------------------------------------------------------------------------

typedef __attribute__((ext_vector_type(16))) __bf16 v16bf;
typedef __attribute__((ext_vector_type(8)))  float   v8f;
typedef __attribute__((ext_vector_type(8)))  __bf16 bf16x8;

#define NVIS   32768
#define CCH    256
#define SSEQ   256
#define NHEAD  8
#define HDIM   32

__device__ __forceinline__ v8f wmma_bf16(v16bf a, v16bf b, v8f c) {
  return __builtin_amdgcn_wmma_f32_16x16x32_bf16(false, a, false, b, (short)0, c,
                                                 false, false);
}

// A operand (16x32, 16-bit): lane holds row (lane&15); K elems:
//   lanes 0-15 : e0..15 = K0..15
//   lanes16-31 : e0..15 = K16..31
__device__ __forceinline__ v16bf load_frag_a(const __bf16* base, int stride,
                                             int k0, int lane) {
  const __bf16* p = base + (lane & 15) * stride + k0 + ((lane >> 4) << 4);
  bf16x8 lo = *(const bf16x8*)p;
  bf16x8 hi = *(const bf16x8*)(p + 8);
  v16bf f;
#pragma unroll
  for (int i = 0; i < 8; ++i) { f[i] = lo[i]; f[i + 8] = hi[i]; }
  return f;
}

// B operand (32x16, 16-bit): source stored as [N][K] row-major (i.e. B^T).
// lane holds column n = lane&15; lanes 0-15: K0..15, lanes 16-31: K16..31.
__device__ __forceinline__ v16bf load_frag_b(const __bf16* base, int stride,
                                             int k0, int lane) {
  const __bf16* p = base + (lane & 15) * stride + k0 + ((lane >> 4) << 4);
  bf16x8 lo = *(const bf16x8*)p;
  bf16x8 hi = *(const bf16x8*)(p + 8);
  v16bf f;
#pragma unroll
  for (int i = 0; i < 8; ++i) { f[i] = lo[i]; f[i + 8] = hi[i]; }
  return f;
}

// B operand from f32 matrix stored [K][N] (k strided): 16 lane-coalesced loads.
__device__ __forceinline__ v16bf load_frag_b_strided_f32(const float* base,
                                                         int stride, int k0,
                                                         int lane) {
  const float* p = base + (size_t)(k0 + ((lane >> 4) << 4)) * stride + (lane & 15);
  v16bf f;
#pragma unroll
  for (int i = 0; i < 16; ++i) f[i] = (__bf16)p[(size_t)i * stride];
  return f;
}

// 3D rope frequency for head_dim=32: zd=yd=10, xd=12.
__device__ __forceinline__ float rope_freq(float dp, float hp, float wp, int hd) {
  float pos, frac;
  if (hd < 10)      { pos = dp; frac = (float)((hd % 5) * 2) * (1.0f / 10.f); }
  else if (hd < 20) { pos = hp; frac = (float)(((hd - 10) % 5) * 2) * (1.0f / 10.f); }
  else              { pos = wp; frac = (float)(((hd - 20) % 6) * 2) * (1.0f / 12.f); }
  return pos * __expf(-9.210340371976184f * frac);  // 10000^{-frac}
}

// ---------------------------------------------------------------------------
// Kernel 0: transpose q_w / o_w into bf16 A-operand matrices [c_out][c_in].
// ---------------------------------------------------------------------------
__global__ void prep_weights_kernel(const float* __restrict__ qw,
                                    const float* __restrict__ ow,
                                    __bf16* __restrict__ qwT,
                                    __bf16* __restrict__ owT) {
  int i = blockIdx.x * 256 + threadIdx.x;  // 65536 total
  int r = i >> 8, c = i & 255;
  qwT[c * 256 + r] = (__bf16)qw[r * 256 + c];
  owT[c * 256 + r] = (__bf16)ow[r * 256 + c];
}

// ---------------------------------------------------------------------------
// Kernel 1: h = gelu_exact(text @ m1_w + m1_b)   [B*S, 256]
// ---------------------------------------------------------------------------
__global__ void text_mlp1_kernel(const float* __restrict__ te,
                                 const float* __restrict__ m1w,
                                 const float* __restrict__ m1b,
                                 float* __restrict__ h) {
  int row = blockIdx.x;   // b*S + s, 512 rows
  int c   = threadIdx.x;  // 256 cols
  const float* t = te + (size_t)row * 512;
  float acc = m1b[c];
  for (int k = 0; k < 512; ++k) acc += t[k] * m1w[k * 256 + c];
  h[(size_t)row * 256 + c] = 0.5f * acc * (1.f + erff(acc * 0.70710678118654752f));
}

// ---------------------------------------------------------------------------
// Kernel 2: K/V linears + phase MLP2 + RoPE(K).
//   k_rot -> [B,NH,S,HD] bf16 (scores B-operand), vT -> [B,NH,HD,S] bf16.
// ---------------------------------------------------------------------------
__global__ void text_kv_kernel(const float* __restrict__ te,
                               const float* __restrict__ kw, const float* __restrict__ kb_,
                               const float* __restrict__ vw, const float* __restrict__ vb_,
                               const float* __restrict__ m2w, const float* __restrict__ m2b,
                               const float* __restrict__ h,
                               __bf16* __restrict__ k_rot, __bf16* __restrict__ vT) {
  __shared__ float kl[256];
  int row = blockIdx.x;         // b*S + s
  int b = row >> 8, s = row & 255;
  int c = threadIdx.x;
  const float* t  = te + (size_t)row * 512;
  const float* hr = h  + (size_t)row * 256;
  float ka = kb_[c], va = vb_[c], pa = m2b[c];
  for (int k = 0; k < 512; ++k) {
    float tv = t[k];
    ka += tv * kw[k * 256 + c];
    va += tv * vw[k * 256 + c];
  }
  for (int k = 0; k < 256; ++k) pa += hr[k] * m2w[k * 256 + c];
  kl[c] = ka;
  __syncthreads();
  int head = c >> 5, hd = c & 31;
  float partner = kl[c ^ 16];
  float rot = (hd < 16) ? -partner : partner;
  float kr = ka * __cosf(pa) + rot * __sinf(pa);
  k_rot[(((size_t)(b * NHEAD + head) * SSEQ + s) << 5) + hd] = (__bf16)kr;
  vT[(((size_t)(b * NHEAD + head) << 5) + hd) * SSEQ + s]    = (__bf16)va;
}

// ---------------------------------------------------------------------------
// Kernel 3: Q projection (q = fv^T @ q_w) + 3D RoPE + 1/sqrt(HD) scale.
//   Wave tile: M=64 (c_out, 2 heads) x N=16 (n). q_rot -> [B,NH,N,HD] bf16.
// ---------------------------------------------------------------------------
__global__ void qproj_rope_kernel(const float* __restrict__ fv,
                                  const __bf16* __restrict__ qwT,
                                  const float* __restrict__ qb,
                                  __bf16* __restrict__ q_rot) {
  const int lane = threadIdx.x & 31;
  const int warp = threadIdx.x >> 5;
  int gw = blockIdx.x * 8 + warp;          // 16384 waves
  int ct = gw & 3;
  int nt = (gw >> 2) & 2047;
  int b  = gw >> 13;
  const int n0 = nt * 16;
  const int c0 = ct * 64;
  const float* fvb = fv + (size_t)b * CCH * NVIS + n0;

  v8f acc[4] = {};
  for (int k0 = 0; k0 < 256; k0 += 32) {
    if (k0 < 224) __builtin_prefetch(fvb + (size_t)(k0 + 32) * NVIS, 0, 3);
    v16bf bfrag = load_frag_b_strided_f32(fvb, NVIS, k0, lane);
#pragma unroll
    for (int st = 0; st < 4; ++st) {
      v16bf afrag = load_frag_a(qwT + (c0 + st * 16) * 256, 256, k0, lane);
      acc[st] = wmma_bf16(afrag, bfrag, acc[st]);
    }
  }

  const int n  = n0 + (lane & 15);
  const float dp = (float)(n >> 10);
  const float hp = (float)((n >> 5) & 31);
  const float wp = (float)(n & 31);
  const float scale = 0.17677669529663687f;  // HD^-0.5
  const int half = lane >> 4;
#pragma unroll
  for (int pr = 0; pr < 2; ++pr) {         // head pair within wave tile
    int head = (c0 >> 5) + pr;
    bf16x8 lopack, hipack;
#pragma unroll
    for (int j = 0; j < 8; ++j) {
      int m = j + 8 * half;                // hd_lo = m, hd_hi = m+16
      float ql = acc[pr * 2 + 0][j] + qb[c0 + pr * 32 + m];
      float qh = acc[pr * 2 + 1][j] + qb[c0 + pr * 32 + 16 + m];
      float fl = rope_freq(dp, hp, wp, m);
      float fh = rope_freq(dp, hp, wp, m + 16);
      lopack[j] = (__bf16)((ql * __cosf(fl) - qh * __sinf(fl)) * scale);
      hipack[j] = (__bf16)((qh * __cosf(fh) + ql * __sinf(fh)) * scale);
    }
    __bf16* dst = q_rot + (((size_t)(b * NHEAD + head) * NVIS + n) << 5);
    *(bf16x8*)(dst + 8 * half)      = lopack;
    *(bf16x8*)(dst + 16 + 8 * half) = hipack;
  }
}

// ---------------------------------------------------------------------------
// Kernel 4: attention. Per wave: 16 q rows x (S=256).
//   scores = Q@K_rot^T (16 WMMAs) -> register softmax -> P@V (16 WMMAs,
//   P transposed to A-layout through wave-private LDS).
// ---------------------------------------------------------------------------
__global__ void attn_kernel(const __bf16* __restrict__ q_rot,
                            const __bf16* __restrict__ k_rot,
                            const __bf16* __restrict__ vT,
                            __bf16* __restrict__ attn_out) {
  __shared__ __attribute__((aligned(16))) __bf16 pbuf[8 * 16 * 32];
  const int lane = threadIdx.x & 31;
  const int warp = threadIdx.x >> 5;
  __bf16* pw = pbuf + warp * (16 * 32);
  int gw = blockIdx.x * 8 + warp;          // 32768 waves
  int nt   = gw & 2047;
  int head = (gw >> 11) & 7;
  int b    = gw >> 14;
  const int n0 = nt * 16;
  const __bf16* qbase = q_rot + (((size_t)(b * NHEAD + head) * NVIS + n0) << 5);
  const __bf16* kbase = k_rot + ((size_t)(b * NHEAD + head) * SSEQ * HDIM);
  const __bf16* vbase = vT    + ((size_t)(b * NHEAD + head) * SSEQ * HDIM);

  v16bf aq = load_frag_a(qbase, 32, 0, lane);

  v8f sc[16];
#pragma unroll
  for (int t = 0; t < 16; ++t) {
    v16bf bk = load_frag_b(kbase + t * 16 * 32, 32, 0, lane);
    v8f z = {};
    sc[t] = wmma_bf16(aq, bk, z);
  }

  // row softmax: rows live in one 16-lane half (M = j + 8*(lane>=16)).
  float mx[8], sm[8];
#pragma unroll
  for (int j = 0; j < 8; ++j) {
    float m = sc[0][j];
#pragma unroll
    for (int t = 1; t < 16; ++t) m = fmaxf(m, sc[t][j]);
    m = fmaxf(m, __shfl_xor(m, 1, 32));
    m = fmaxf(m, __shfl_xor(m, 2, 32));
    m = fmaxf(m, __shfl_xor(m, 4, 32));
    m = fmaxf(m, __shfl_xor(m, 8, 32));
    mx[j] = m;
  }
#pragma unroll
  for (int j = 0; j < 8; ++j) {
    float s = 0.f;
#pragma unroll
    for (int t = 0; t < 16; ++t) {
      float e = __expf(sc[t][j] - mx[j]);
      sc[t][j] = e;
      s += e;
    }
    s += __shfl_xor(s, 1, 32);
    s += __shfl_xor(s, 2, 32);
    s += __shfl_xor(s, 4, 32);
    s += __shfl_xor(s, 8, 32);
    sm[j] = s;
  }

  // P @ V, K chunks of 32 (two score tiles per chunk).
  v8f out0 = {}, out1 = {};
  const int half = lane >> 4;
#pragma unroll
  for (int kc = 0; kc < 8; ++kc) {
#pragma unroll
    for (int j = 0; j < 8; ++j) {
      int m = j + 8 * half;
      pw[m * 32 + (lane & 15)]      = (__bf16)sc[2 * kc][j];
      pw[m * 32 + 16 + (lane & 15)] = (__bf16)sc[2 * kc + 1][j];
    }
    asm volatile("s_wait_dscnt 0x0" ::: "memory");
    v16bf ap  = load_frag_a(pw, 32, 0, lane);
    v16bf bv0 = load_frag_b(vbase,            SSEQ, kc * 32, lane);
    v16bf bv1 = load_frag_b(vbase + 16 * SSEQ, SSEQ, kc * 32, lane);
    out0 = wmma_bf16(ap, bv0, out0);
    out1 = wmma_bf16(ap, bv1, out1);
    asm volatile("" ::: "memory");  // keep LDS reads ahead of next writes
  }

  // normalize + store [B, N, C] bf16.
#pragma unroll
  for (int j = 0; j < 8; ++j) {
    int m = j + 8 * half;
    int n = n0 + m;
    float r = 1.f / sm[j];
    __bf16* dst = attn_out + ((size_t)(b * NVIS + n) << 8) + head * 32 + (lane & 15);
    dst[0]  = (__bf16)(out0[j] * r);
    dst[16] = (__bf16)(out1[j] * r);
  }
}

// ---------------------------------------------------------------------------
// Kernel 5: output projection, stores transposed f32 result [B,C,D,H,W].
// ---------------------------------------------------------------------------
__global__ void oproj_kernel(const __bf16* __restrict__ attn_out,
                             const __bf16* __restrict__ owT,
                             const float* __restrict__ ob,
                             float* __restrict__ out) {
  const int lane = threadIdx.x & 31;
  const int warp = threadIdx.x >> 5;
  int gw = blockIdx.x * 8 + warp;          // 16384 waves
  int ct = gw & 3;
  int nt = (gw >> 2) & 2047;
  int b  = gw >> 13;
  const int n0 = nt * 16, c0 = ct * 64;
  const __bf16* ab = attn_out + ((size_t)(b * NVIS + n0) << 8);

  v8f acc[4] = {};
  for (int k0 = 0; k0 < 256; k0 += 32) {
    v16bf bfrag = load_frag_b(ab, 256, k0, lane);
#pragma unroll
    for (int st = 0; st < 4; ++st) {
      v16bf afrag = load_frag_a(owT + (c0 + st * 16) * 256, 256, k0, lane);
      acc[st] = wmma_bf16(afrag, bfrag, acc[st]);
    }
  }
  const int half = lane >> 4;
  const int n = n0 + (lane & 15);
#pragma unroll
  for (int st = 0; st < 4; ++st) {
#pragma unroll
    for (int j = 0; j < 8; ++j) {
      int c = c0 + st * 16 + j + 8 * half;
      out[((size_t)(b * CCH + c) << 15) + n] = acc[st][j] + ob[c];
    }
  }
}

// ---------------------------------------------------------------------------
extern "C" void kernel_launch(void* const* d_in, const int* in_sizes, int n_in,
                              void* d_out, int out_size, void* d_ws, size_t ws_size,
                              hipStream_t stream) {
  (void)in_sizes; (void)n_in; (void)out_size; (void)ws_size;
  const float* fv  = (const float*)d_in[0];
  const float* te  = (const float*)d_in[1];
  const float* qw  = (const float*)d_in[2];
  const float* qb  = (const float*)d_in[3];
  const float* kw  = (const float*)d_in[4];
  const float* kb  = (const float*)d_in[5];
  const float* vw  = (const float*)d_in[6];
  const float* vb  = (const float*)d_in[7];
  const float* ow  = (const float*)d_in[8];
  const float* ob  = (const float*)d_in[9];
  const float* m1w = (const float*)d_in[10];
  const float* m1b = (const float*)d_in[11];
  const float* m2w = (const float*)d_in[12];
  const float* m2b = (const float*)d_in[13];
  float* out = (float*)d_out;

  char* ws = (char*)d_ws;
  size_t off = 0;
  __bf16* qwT   = (__bf16*)(ws + off); off += (size_t)256 * 256 * 2;        // 128 KB
  __bf16* owT   = (__bf16*)(ws + off); off += (size_t)256 * 256 * 2;        // 128 KB
  float*  h     = (float*) (ws + off); off += (size_t)2 * 256 * 256 * 4;    // 512 KB
  __bf16* k_rot = (__bf16*)(ws + off); off += (size_t)2 * 8 * 256 * 32 * 2; // 256 KB
  __bf16* vT    = (__bf16*)(ws + off); off += (size_t)2 * 8 * 256 * 32 * 2; // 256 KB
  __bf16* q_rot = (__bf16*)(ws + off); off += (size_t)2 * 8 * 32768 * 32 * 2; // 32 MB
  __bf16* a_out = (__bf16*)(ws + off); off += (size_t)2 * 32768 * 256 * 2;    // 32 MB

  prep_weights_kernel<<<256, 256, 0, stream>>>(qw, ow, qwT, owT);
  text_mlp1_kernel<<<512, 256, 0, stream>>>(te, m1w, m1b, h);
  text_kv_kernel<<<512, 256, 0, stream>>>(te, kw, kb, vw, vb, m2w, m2b, h, k_rot, vT);
  qproj_rope_kernel<<<2048, 256, 0, stream>>>(fv, qwT, qb, q_rot);
  attn_kernel<<<4096, 256, 0, stream>>>(q_rot, k_rot, vT, a_out);
  oproj_kernel<<<2048, 256, 0, stream>>>(a_out, owT, ob, out);
}
